// GAT_47184510714009
// MI455X (gfx1250) — matrix operand order repaired and checked by reading, not compiled, not run on hardware
//
#include <hip/hip_runtime.h>
#include <hip/hip_bf16.h>
#include <stdint.h>

typedef __attribute__((ext_vector_type(16))) __bf16 v16bf;
typedef __attribute__((ext_vector_type(8)))  float  v8f;
typedef __attribute__((ext_vector_type(4)))  unsigned int u32x4;
typedef __attribute__((ext_vector_type(8)))  int   i32x8;
typedef __attribute__((ext_vector_type(4)))  int   i32x4;

#define ALPHA 0.2f
constexpr int Bn = 4, Hn = 4, Nn = 2048, Fn = 128, Dn = 32;
constexpr int BH = Bn * Hn;
constexpr int CH = 256;            // j-chunk staged per TDM transfer
constexpr int NCH = Nn / CH;       // 8 chunks

__device__ __forceinline__ float leaky(float x) { return x > 0.f ? x : ALPHA * x; }

// ---------------------------------------------------------------------------
// TDM: one DMA pulls a 32 x CH bf16 tile (row stride Nn elements) into LDS.
// D# per CDNA5 ISA: group0 {count, lds_addr, gaddr, type=2},
// group1 {data_size=2B, tensor_dim0=CH, tensor_dim1=32, tile_dim0=CH,
//         tile_dim1=32, tensor_dim0_stride=Nn}.  6-arg builtin on this toolchain.
// ---------------------------------------------------------------------------
__device__ __forceinline__ void tdm_load_tile(const __bf16* gsrc, uint32_t lds_off)
{
    const uint64_t ga = (uint64_t)(uintptr_t)gsrc;
    u32x4 g0;
    g0[0] = 1u;                                                // count=1 (user D#)
    g0[1] = lds_off;                                           // lds_addr (bytes)
    g0[2] = (uint32_t)(ga & 0xFFFFFFFFu);                      // global_addr[31:0]
    g0[3] = (uint32_t)((ga >> 32) & 0x1FFFFFFu) | (2u << 30);  // addr[56:32]|type=2
    i32x8 g1;
    g1[0] = (int)(1u << 16);            // data_size=1 -> 2 bytes; no flags
    g1[1] = (int)((unsigned)CH << 16);  // tensor_dim0[15:0] @ bits 63:48
    g1[2] = (int)(32u << 16);           // tensor_dim1[15:0] @ bits 95:80
    g1[3] = (int)((unsigned)CH << 16);  // tile_dim0 @ bits 127:112
    g1[4] = 32;                         // tile_dim1 @ bits 143:128 (tile_dim2=0)
    g1[5] = Nn;                         // tensor_dim0_stride[31:0]
    g1[6] = 0;                          // stride0[47:32] | dim1_stride lo
    g1[7] = 0;
    i32x4 gz4 = {0, 0, 0, 0};
    i32x8 gz8 = {0, 0, 0, 0, 0, 0, 0, 0};
    __builtin_amdgcn_tensor_load_to_lds(g0, g1, gz4, gz4, gz8, 0);
}

// ---------------------------------------------------------------------------
// Kernel 1: Wh = x[b] @ W[h]  via bf16 WMMA (f32 accum).
// Outputs: WhT bf16 [bh][d][n] (d-major for contiguous B-fragments later),
//          s[bh][n] = Wh . a1,  t[bh][n] = Wh . a2.
// ---------------------------------------------------------------------------
__global__ __launch_bounds__(256) void gat_proj(const float* __restrict__ x,
                                                const float* __restrict__ W,
                                                const float* __restrict__ a,
                                                __bf16* __restrict__ WhT,
                                                float* __restrict__ s,
                                                float* __restrict__ t)
{
    __shared__ __align__(32) __bf16 lWT[Dn * Fn]; // W[h]^T : [d][k] bf16, 8 KB
    const int bh   = blockIdx.x >> 4;
    const int band = blockIdx.x & 15;
    const int b = bh >> 2, h = bh & 3;
    const int tid  = threadIdx.x;
    const int lane = tid & 31, wave = tid >> 5;
    const int lm = lane & 15, lh = lane >> 4;

    #pragma unroll
    for (int q = 0; q < 16; ++q) {
        int e = tid * 16 + q;
        int k = e >> 5, d = e & 31;
        lWT[d * Fn + k] = (__bf16)W[(h * Fn + k) * Dn + d];
    }
    __syncthreads();

    const int i0 = band * 128 + wave * 16;
    v8f acc0 = {}, acc1 = {};
    #pragma unroll
    for (int kc = 0; kc < Fn; kc += 32) {
        const float* xr = x + ((size_t)b * Nn + (i0 + lm)) * Fn + kc + lh * 16;
        float4 s0 = *reinterpret_cast<const float4*>(xr);
        float4 s1 = *reinterpret_cast<const float4*>(xr + 4);
        float4 s2 = *reinterpret_cast<const float4*>(xr + 8);
        float4 s3 = *reinterpret_cast<const float4*>(xr + 12);
        v16bf av;
        av[0]=(__bf16)s0.x; av[1]=(__bf16)s0.y; av[2]=(__bf16)s0.z; av[3]=(__bf16)s0.w;
        av[4]=(__bf16)s1.x; av[5]=(__bf16)s1.y; av[6]=(__bf16)s1.z; av[7]=(__bf16)s1.w;
        av[8]=(__bf16)s2.x; av[9]=(__bf16)s2.y; av[10]=(__bf16)s2.z; av[11]=(__bf16)s2.w;
        av[12]=(__bf16)s3.x; av[13]=(__bf16)s3.y; av[14]=(__bf16)s3.z; av[15]=(__bf16)s3.w;
        v16bf b0 = *reinterpret_cast<const v16bf*>(&lWT[(lm)      * Fn + kc + lh * 16]);
        v16bf b1 = *reinterpret_cast<const v16bf*>(&lWT[(16 + lm) * Fn + kc + lh * 16]);
        acc0 = __builtin_amdgcn_wmma_f32_16x16x32_bf16(false, av, false, b0, (short)0, acc0, false, false);
        acc1 = __builtin_amdgcn_wmma_f32_16x16x32_bf16(false, av, false, b1, (short)0, acc1, false, false);
    }

    const float a1lo = a[h * 2 * Dn + lm],      a1hi = a[h * 2 * Dn + 16 + lm];
    const float a2lo = a[h * 2 * Dn + Dn + lm], a2hi = a[h * 2 * Dn + Dn + 16 + lm];
    float sp[8], tp[8];
    #pragma unroll
    for (int r = 0; r < 8; ++r) {
        sp[r] = acc0[r] * a1lo + acc1[r] * a1hi;
        tp[r] = acc0[r] * a2lo + acc1[r] * a2hi;
    }
    #pragma unroll
    for (int off = 1; off < 16; off <<= 1) {
        #pragma unroll
        for (int r = 0; r < 8; ++r) {
            sp[r] += __shfl_xor(sp[r], off);
            tp[r] += __shfl_xor(tp[r], off);
        }
    }
    if (lm == 0) {
        #pragma unroll
        for (int r = 0; r < 8; ++r) {
            int row = i0 + lh * 8 + r;
            s[(size_t)bh * Nn + row] = sp[r];
            t[(size_t)bh * Nn + row] = tp[r];
        }
    }
    #pragma unroll
    for (int r = 0; r < 8; ++r) {
        int row = i0 + lh * 8 + r;
        WhT[((size_t)bh * Dn + lm)      * Nn + row] = (__bf16)acc0[r];
        WhT[((size_t)bh * Dn + 16 + lm) * Nn + row] = (__bf16)acc1[r];
    }
}

// ---------------------------------------------------------------------------
// Kernel 1b: T[bh] = max_j t[bh][j]  (leakyrelu monotone => softmax max free).
// ---------------------------------------------------------------------------
__global__ __launch_bounds__(256) void gat_tmax(const float* __restrict__ t,
                                                float* __restrict__ T)
{
    __shared__ float red[256];
    const int bh = blockIdx.x, tid = threadIdx.x;
    float m = -1e30f;
    for (int j = tid; j < Nn; j += 256) m = fmaxf(m, t[(size_t)bh * Nn + j]);
    red[tid] = m;
    __syncthreads();
    for (int off = 128; off > 0; off >>= 1) {
        if (tid < off) red[tid] = fmaxf(red[tid], red[tid + off]);
        __syncthreads();
    }
    if (tid == 0) T[bh] = red[0];
}

// ---------------------------------------------------------------------------
// Kernel 2: fused softmax(attn) @ Wh, flash-style single pass.
// Wh^T chunks arrive via double-buffered TENSOR_LOAD_TO_LDS (wave 0 issues,
// waits TENSORcnt, workgroup barrier publishes). P tiles built in registers
// (v_exp_f32 -> bf16), contracted on v_wmma_f32_16x16x32_bf16.
// ---------------------------------------------------------------------------
__global__ __launch_bounds__(256) void gat_attn(const __bf16* __restrict__ WhT,
                                                const float* __restrict__ s,
                                                const float* __restrict__ t,
                                                const float* __restrict__ T,
                                                float* __restrict__ out)
{
    __shared__ __align__(32) __bf16 lB[2][Dn * CH]; // 2 x 16 KB Wh^T tiles
    __shared__ float lt[2][CH];
    const int bh = blockIdx.x >> 4, band = blockIdx.x & 15;
    const int b = bh >> 2, h = bh & 3;
    const int tid = threadIdx.x, lane = tid & 31, wave = tid >> 5;
    const int lm = lane & 15, lh = lane >> 4;
    const int i0 = band * 128 + wave * 16;

    const float sl = s[(size_t)bh * Nn + i0 + lm];
    const float ml = leaky(sl + T[bh]);

    const __bf16* gW = WhT + (size_t)bh * Dn * Nn;

    v8f acc0 = {}, acc1 = {};
    float rsum = 0.f;

    // Prologue: stage chunk 0 into buffer 0
    if (wave == 0) tdm_load_tile(gW, (uint32_t)(uintptr_t)&lB[0][0]);
    lt[0][tid] = t[(size_t)bh * Nn + tid];
    if (wave == 0) __builtin_amdgcn_s_wait_tensorcnt(0);
    __syncthreads();

    for (int c = 0; c < NCH; ++c) {
        const int cur = c & 1, nxt = cur ^ 1;
        if (c + 1 < NCH) {   // kick off next chunk while computing this one
            if (wave == 0) tdm_load_tile(gW + (c + 1) * CH,
                                         (uint32_t)(uintptr_t)&lB[nxt][0]);
            lt[nxt][tid] = t[(size_t)bh * Nn + (c + 1) * CH + tid];
        }

        #pragma unroll
        for (int jb = 0; jb < CH; jb += 32) {
            v16bf av;
            #pragma unroll
            for (int q = 0; q < 8; ++q) {   // K = lh*16 + q
                float p = __expf(leaky(sl + lt[cur][jb + lh * 16 + q]) - ml);
                rsum += p;
                av[q] = (__bf16)p;
            }
            #pragma unroll
            for (int q = 0; q < 8; ++q) {   // K = lh*16 + 8 + q
                float p = __expf(leaky(sl + lt[cur][jb + lh * 16 + 8 + q]) - ml);
                rsum += p;
                av[8 + q] = (__bf16)p;
            }
            v16bf b0 = *reinterpret_cast<const v16bf*>(&lB[cur][(lm)      * CH + jb + lh * 16]);
            v16bf b1 = *reinterpret_cast<const v16bf*>(&lB[cur][(16 + lm) * CH + jb + lh * 16]);
            acc0 = __builtin_amdgcn_wmma_f32_16x16x32_bf16(false, av, false, b0, (short)0, acc0, false, false);
            acc1 = __builtin_amdgcn_wmma_f32_16x16x32_bf16(false, av, false, b1, (short)0, acc1, false, false);
        }

        if (wave == 0 && c + 1 < NCH) __builtin_amdgcn_s_wait_tensorcnt(0);
        __syncthreads();
    }

    // Full row sums: row m lives in lanes m and m+16 (disjoint K halves).
    rsum += __shfl_xor(rsum, 16);

    #pragma unroll
    for (int r = 0; r < 8; ++r) {
        const int row = lh * 8 + r;            // C/D layout: VGPR r -> M = r + 8*half
        const float denom = __shfl(rsum, row);
        const int i = i0 + row;
        float v0 = acc0[r] / denom;
        float v1 = acc1[r] / denom;
        v0 = v0 > 0.f ? v0 : __expf(v0) - 1.f;  // ELU
        v1 = v1 > 0.f ? v1 : __expf(v1) - 1.f;
        out[((size_t)b * Nn + i) * (Hn * Dn) + h * Dn + lm]      = v0;
        out[((size_t)b * Nn + i) * (Hn * Dn) + h * Dn + 16 + lm] = v1;
    }
}

// ---------------------------------------------------------------------------
extern "C" void kernel_launch(void* const* d_in, const int* in_sizes, int n_in,
                              void* d_out, int out_size, void* d_ws, size_t ws_size,
                              hipStream_t stream)
{
    const float* x = (const float*)d_in[0];   // (4, 2048, 128)
    const float* W = (const float*)d_in[1];   // (4, 128, 32)
    const float* a = (const float*)d_in[2];   // (4, 64)
    float* out = (float*)d_out;               // (4, 2048, 128)

    char* ws = (char*)d_ws;
    constexpr size_t whtBytes = (size_t)BH * Dn * Nn * sizeof(__bf16); // 2 MB
    constexpr size_t stBytes  = (size_t)BH * Nn * sizeof(float);       // 128 KB
    __bf16* WhT = (__bf16*)ws;
    float*  s   = (float*)(ws + whtBytes);
    float*  t   = (float*)(ws + whtBytes + stBytes);
    float*  T   = (float*)(ws + whtBytes + 2 * stBytes);

    gat_proj<<<dim3(BH * 16), dim3(256), 0, stream>>>(x, W, a, WhT, s, t);
    gat_tmax<<<dim3(BH),      dim3(256), 0, stream>>>(t, T);
    gat_attn<<<dim3(BH * 16), dim3(256), 0, stream>>>(WhT, s, t, T, out);
}
